// MambaNet_3238405341499
// MI455X (gfx1250) — matrix operand order repaired and checked by reading, not compiled, not run on hardware
//
#include <hip/hip_runtime.h>
#include <hip/hip_bf16.h>

typedef __attribute__((ext_vector_type(16))) _Float16 v16h;
typedef __attribute__((ext_vector_type(8)))  _Float16 v8h;
typedef __attribute__((ext_vector_type(8)))  float    v8f;

typedef int v4i __attribute__((vector_size(16)));
typedef __attribute__((address_space(1))) v4i* gv4p;
typedef __attribute__((address_space(3))) v4i* lv4p;

#if defined(__AMDGCN__) && __has_builtin(__builtin_amdgcn_global_load_async_to_lds_b128)
#define GEMM_ASYNC 1
#else
#define GEMM_ASYNC 0
#endif

__device__ inline gv4p to_g(const void* p) { return __builtin_bit_cast(gv4p, p); }
__device__ inline lv4p to_l(void* p) { return (lv4p)(uint32_t)(uintptr_t)p; }

__device__ inline void wait_async0() {
#if GEMM_ASYNC
#if __has_builtin(__builtin_amdgcn_s_wait_asynccnt)
    __builtin_amdgcn_s_wait_asynccnt(0);
#else
    asm volatile("s_wait_asynccnt 0x0" ::: "memory");
#endif
#endif
}

// ---------------------------------------------------------------- helpers
__device__ inline float gelu_f(float x) {
    const float c = 0.7978845608028654f;
    float x3 = x * x * x;
    return 0.5f * x * (1.0f + tanhf(c * (x + 0.044715f * x3)));
}
__device__ inline float softplus_f(float x) {
    return x > 20.0f ? x : log1pf(expf(x));
}

// ---------------------------------------------------------------- zero a small f16 region (zero-pad source rows)
__global__ void zero_kernel(_Float16* __restrict__ p, int n) {
    int i = blockIdx.x * 256 + threadIdx.x;
    if (i < n) p[i] = (_Float16)0.0f;
}

// ---------------------------------------------------------------- f32 -> f16
__global__ void cvt_f16_kernel(const float* __restrict__ s, _Float16* __restrict__ d, int n) {
    int i = blockIdx.x * 256 + threadIdx.x;
    if (i < n) d[i] = (_Float16)s[i];
}

// ---------------------------------------------------------------- elementwise add
__global__ void add_kernel(const float* __restrict__ a, const float* __restrict__ b,
                           float* __restrict__ o, int n) {
    int i = blockIdx.x * 256 + threadIdx.x;
    if (i < n) o[i] = a[i] + b[i];
}

// ---------------------------------------------------------------- NCHW (4,512,16,16) -> seq [l*4+b][c]
__global__ void toseq_kernel(const float* __restrict__ src, float* __restrict__ seq) {
    int id = blockIdx.x;            // 2048 = 4*512
    int b = id >> 9, c = id & 511;
    const float* s = src + ((size_t)b * 512 + c) * 256;
    int l = threadIdx.x;            // 256 threads
    seq[((size_t)(l * 4 + b)) * 512 + c] = s[l];
}

// ---------------------------------------------------------------- seq -> output NCHW
__global__ void outtr_kernel(const float* __restrict__ seq, float* __restrict__ out) {
    int id = blockIdx.x;            // 2048 = 4*512
    int b = id >> 9, c = id & 511;
    int l = threadIdx.x;
    out[((size_t)b * 512 + c) * 256 + l] = seq[((size_t)(l * 4 + b)) * 512 + c];
}

// ---------------------------------------------------------------- adaptive pool 16x16 -> 18x18 (f16 out for GEMM A)
__global__ void pool_kernel(const float* __restrict__ content, _Float16* __restrict__ pooled_h) {
    int bp = blockIdx.x;            // 1296 = 4*324
    int b = bp / 324, p = bp % 324;
    int i = p / 18, j = p % 18;
    int s0 = (i * 16) / 18, e0 = ((i + 1) * 16 + 17) / 18;
    int s1 = (j * 16) / 18, e1 = ((j + 1) * 16 + 17) / 18;
    float scale = 1.0f / (float)((e0 - s0) * (e1 - s1));
    for (int c = threadIdx.x; c < 512; c += 256) {
        const float* base = content + ((size_t)b * 512 + c) * 256;
        float acc = 0.0f;
        for (int h = s0; h < e0; ++h)
            for (int w = s1; w < e1; ++w)
                acc += base[h * 16 + w];
        pooled_h[(size_t)bp * 512 + c] = (_Float16)(acc * scale);
    }
}

// ---------------------------------------------------------------- bilinear 18x18 -> 16x16, emit seq layout
__global__ void resize_kernel(const float* __restrict__ posg, float* __restrict__ pc_seq) {
    int id = blockIdx.x;            // 1024 = 256*4
    int b = id & 3, l = id >> 2;
    int y = l >> 4, x = l & 15;
    float fy = 1.125f * (float)y + 0.0625f;
    float fx = 1.125f * (float)x + 0.0625f;
    int iy = (int)floorf(fy), ix = (int)floorf(fx);
    float wy = fy - (float)iy, wx = fx - (float)ix;
    int iy1 = min(iy + 1, 17), ix1 = min(ix + 1, 17);
    for (int o = threadIdx.x; o < 512; o += 256) {
        const float* gb = posg + o;
        auto g = [&](int yy, int xx) {
            return gb[((size_t)(b * 324 + yy * 18 + xx)) * 512];
        };
        float v = (1.f - wy) * ((1.f - wx) * g(iy, ix)  + wx * g(iy, ix1)) +
                  wy         * ((1.f - wx) * g(iy1, ix) + wx * g(iy1, ix1));
        pc_seq[((size_t)(l * 4 + b)) * 512 + o] = v;
    }
}

// ---------------------------------------------------------------- layernorm over 512, optional add/accumulate, f32/f16 outs
__global__ void ln_kernel(const float* __restrict__ x, const float* __restrict__ add,
                          const float* __restrict__ g, const float* __restrict__ b,
                          float* outf, _Float16* outh, int accumulate) {
    int row = blockIdx.x;
    int t = threadIdx.x;            // 256 threads, 2 channels each
    size_t base = (size_t)row * 512;
    float v0 = x[base + t], v1 = x[base + 256 + t];
    if (add) { v0 += add[base + t]; v1 += add[base + 256 + t]; }
    float s = v0 + v1, sq = v0 * v0 + v1 * v1;
    for (int o = 16; o > 0; o >>= 1) {
        s  += __shfl_xor(s,  o, 32);
        sq += __shfl_xor(sq, o, 32);
    }
    __shared__ float ps[8], pq[8];
    if ((t & 31) == 0) { ps[t >> 5] = s; pq[t >> 5] = sq; }
    __syncthreads();
    float S = 0.f, Q = 0.f;
    for (int i = 0; i < 8; ++i) { S += ps[i]; Q += pq[i]; }
    float mean = S * (1.0f / 512.0f);
    float var  = Q * (1.0f / 512.0f) - mean * mean;
    float inv  = rsqrtf(var + 1e-5f);
    float r0 = (v0 - mean) * inv * g[t]       + b[t];
    float r1 = (v1 - mean) * inv * g[t + 256] + b[t + 256];
    if (accumulate) { r0 += outf[base + t]; r1 += outf[base + 256 + t]; }
    if (outf) { outf[base + t] = r0; outf[base + 256 + t] = r1; }
    if (outh) { outh[base + t] = (_Float16)r0; outh[base + 256 + t] = (_Float16)r1; }
}

// ---------------------------------------------------------------- depthwise conv (along H=4, kw=1 column) + gelu
__global__ void conv_gelu_kernel(const float* __restrict__ u, const float* __restrict__ cw,
                                 const float* __restrict__ cb, float* __restrict__ v,
                                 _Float16* __restrict__ vh, _Float16* __restrict__ vfh) {
    int l = blockIdx.x;             // 256 token groups of 4 rows
    int t = threadIdx.x;
    for (int cc = 0; cc < 2; ++cc) {
        int c = t + cc * 256;
        float w0 = cw[(c * 3 + 0) * 3 + 1];
        float w1 = cw[(c * 3 + 1) * 3 + 1];
        float w2 = cw[(c * 3 + 2) * 3 + 1];
        float bias = cb[c];
        float uv[4];
        for (int h = 0; h < 4; ++h) uv[h] = u[((size_t)(l * 4 + h)) * 512 + c];
        for (int h = 0; h < 4; ++h) {
            float acc = bias + w1 * uv[h];
            if (h > 0) acc += w0 * uv[h - 1];
            if (h < 3) acc += w2 * uv[h + 1];
            float gv = gelu_f(acc);
            v[((size_t)(l * 4 + h)) * 512 + c] = gv;
            vh[((size_t)(l * 4 + h)) * 512 + c] = (_Float16)gv;
            vfh[((size_t)(l * 4 + (3 - h))) * 512 + c] = (_Float16)gv;
        }
    }
}

// ---------------------------------------------------------------- WMMA GEMM: out[M,N] = A[M,K](f16) @ W[N,K]^T(f16) (+bias,+gelu,+residual)
// Double-buffered LDS, one barrier per K-step, branchless OOB via zero-pad rows,
// async global->LDS staging when available.
__global__ __launch_bounds__(256) void gemm_kernel(
    const _Float16* __restrict__ A, const _Float16* __restrict__ W,
    const float* __restrict__ bias, const float* __restrict__ residual,
    float* __restrict__ outf, _Float16* __restrict__ outh,
    const _Float16* __restrict__ zpad,
    int M, int N, int K, int act) {
    __shared__ __align__(16) _Float16 As[2][64][40];
    __shared__ __align__(16) _Float16 Bs[2][64][40];
    const int mBase = blockIdx.x * 64;
    const int nBase = blockIdx.y * 64;
    const int t = threadIdx.x;
    const int lane = t & 31;
    const int w = t >> 5;
    const int wm = w & 3;           // 0..3 -> 16-row slice
    const int wn = w >> 2;          // 0..1 -> 32-col slice
    const int lrow = t >> 2;        // 0..63, loader row
    const int lk = (t & 3) * 8;     // loader k offset (halfs)
    const int gm = mBase + lrow;
    const int gn = nBase + lrow;

    // branchless OOB: out-of-range rows stream zeros from zpad with stride 0
    const _Float16* asrc = (gm < M) ? (A + (size_t)gm * K + lk) : zpad;
    const _Float16* bsrc = (gn < N) ? (W + (size_t)gn * K + lk) : zpad;
    const int astep = (gm < M) ? 32 : 0;
    const int bstep = (gn < N) ? 32 : 0;

    v8f acc0 = {};
    v8f acc1 = {};

    auto issue = [&](int p) {
#if GEMM_ASYNC
        __builtin_amdgcn_global_load_async_to_lds_b128(to_g(asrc), to_l(&As[p][lrow][lk]), 0, 0);
        __builtin_amdgcn_global_load_async_to_lds_b128(to_g(bsrc), to_l(&Bs[p][lrow][lk]), 0, 0);
#else
        *(v8h*)&As[p][lrow][lk] = *(const v8h*)asrc;
        *(v8h*)&Bs[p][lrow][lk] = *(const v8h*)bsrc;
        __builtin_prefetch(asrc + astep, 0, 1);
        __builtin_prefetch(bsrc + bstep, 0, 1);
#endif
        asrc += astep;
        bsrc += bstep;
    };

    issue(0);
    wait_async0();
    __syncthreads();

    const int nIter = K >> 5;
    int p = 0;
    for (int it = 0; it < nIter; ++it) {
        if (it + 1 < nIter) issue(p ^ 1);

        // A fragment: lane<16 -> row m=lane, K {0..7,16..23}; lane>=16 -> K {8..15,24..31}
        const _Float16* Ap = &As[p][16 * wm + (lane & 15)][0];
        int kH = (lane >> 4) << 3;
        v8h alo = *(const v8h*)(Ap + kH);
        v8h ahi = *(const v8h*)(Ap + kH + 16);
        v16h af = __builtin_shufflevector(alo, ahi, 0,1,2,3,4,5,6,7,8,9,10,11,12,13,14,15);

        // B fragments: lane selects N=lane&15, K same as A: {0..7,16..23} / {8..15,24..31}
        int koff = (lane >> 4) << 3;
        const _Float16* Bp0 = &Bs[p][32 * wn + (lane & 15)][koff];
        const _Float16* Bp1 = &Bs[p][32 * wn + 16 + (lane & 15)][koff];
        v8h b0l = *(const v8h*)Bp0, b0h = *(const v8h*)(Bp0 + 16);
        v8h b1l = *(const v8h*)Bp1, b1h = *(const v8h*)(Bp1 + 16);
        v16h bf0 = __builtin_shufflevector(b0l, b0h, 0,1,2,3,4,5,6,7,8,9,10,11,12,13,14,15);
        v16h bf1 = __builtin_shufflevector(b1l, b1h, 0,1,2,3,4,5,6,7,8,9,10,11,12,13,14,15);

        acc0 = __builtin_amdgcn_wmma_f32_16x16x32_f16(false, af, false, bf0, (short)0, acc0, false, false);
        acc1 = __builtin_amdgcn_wmma_f32_16x16x32_f16(false, af, false, bf1, (short)0, acc1, false, false);

        wait_async0();      // next tile landed in LDS (async path)
        __syncthreads();    // everyone done reading buf p; buf p^1 visible
        p ^= 1;
    }

    // epilogue: VGPR r -> M = r + 8*(lane>=16); N = lane&15 within 16-tile
    int mTop = mBase + 16 * wm + ((lane >> 4) << 3);
    int nCol = nBase + 32 * wn + (lane & 15);
    for (int r = 0; r < 8; ++r) {
        int m = mTop + r;
        if (m >= M) break;
        for (int tt = 0; tt < 2; ++tt) {
            int n = nCol + 16 * tt;
            if (n >= N) continue;
            float val = tt ? acc1[r] : acc0[r];
            if (bias) val += bias[n];
            if (act) val = gelu_f(val);
            if (residual) val += residual[(size_t)m * N + n];
            size_t oi = (size_t)m * N + n;
            if (outf) outf[oi] = val;
            if (outh) outh[oi] = (_Float16)val;
        }
    }
}

// ---------------------------------------------------------------- selective scan (L=4), algebraically collapsed; one wave per (bt,k,head)
__global__ void scan_kernel(const float* __restrict__ xdbl01, const float* __restrict__ xdbl23,
                            const float* __restrict__ v, const float* __restrict__ dtb,
                            const float* __restrict__ alog, const float* __restrict__ dvec,
                            float* __restrict__ ys) {
    int wid = (blockIdx.x << 3) + (threadIdx.x >> 5);   // 8192 waves
    int lane = threadIdx.x & 31;
    int h = wid & 7;
    int k = (wid >> 3) & 3;
    int bt = wid >> 5;
    const float* xd = (k < 2) ? xdbl01 : xdbl23;
    int colBase = (k & 1) * 136;
    float A  = -expf(alog[k * 8 + h]);
    float Dv = dvec[k * 8 + h];
    float dtbv = dtb[k * 8 + h];

    float dt[4], pre[4], B0[4], B1[4], C0[4], C1[4];
    for (int s = 0; s < 4; ++s) {
        const float* rowp = xd + (size_t)(bt * 4 + s) * 272 + colBase;
        float d = softplus_f(rowp[h] + dtbv);
        dt[s] = d;
        pre[s] = (s ? pre[s - 1] : 0.0f) + d;
        B0[s] = rowp[8 + lane];       B1[s] = rowp[8 + 32 + lane];
        C0[s] = rowp[72 + lane];      C1[s] = rowp[72 + 32 + lane];
    }
    // G[l][t] = exp(A*(pre_l - pre_t)) * dt_t * (B_t . C_l)
    float G[4][4];
    for (int l = 0; l < 4; ++l)
        for (int s = 0; s <= l; ++s) {
            float p = B0[s] * C0[l] + B1[s] * C1[l];
            for (int o = 16; o > 0; o >>= 1) p += __shfl_xor(p, o, 32);
            G[l][s] = expf(A * (pre[l] - pre[s])) * dt[s] * p;
        }
    float xv[4][2];
    for (int s = 0; s < 4; ++s) {
        int lsrc = (k < 2) ? s : (3 - s);
        const float* vp = v + (size_t)(bt * 4 + lsrc) * 512 + h * 64;
        xv[s][0] = vp[lane];
        xv[s][1] = vp[lane + 32];
    }
    float* out = ys + ((size_t)k * 1024 + bt * 4) * 512 + h * 64;
    for (int l = 0; l < 4; ++l) {
        float y0 = Dv * xv[l][0];
        float y1 = Dv * xv[l][1];
        for (int s = 0; s <= l; ++s) {
            y0 += G[l][s] * xv[s][0];
            y1 += G[l][s] * xv[s][1];
        }
        out[(size_t)l * 512 + lane] = y0;
        out[(size_t)l * 512 + 32 + lane] = y1;
    }
}

// ---------------------------------------------------------------- directional combine: y = ys0 + ys1 + flip(ys2) + flip(ys3)
__global__ void combine_kernel(const float* __restrict__ ys, float* __restrict__ y) {
    int idx = blockIdx.x * 256 + threadIdx.x;   // 1024*512
    int m = idx >> 9, c = idx & 511;
    int mf = (m & ~3) | (3 - (m & 3));
    y[idx] = ys[idx] + ys[524288 + idx] +
             ys[2 * 524288 + (size_t)mf * 512 + c] +
             ys[3 * 524288 + (size_t)mf * 512 + c];
}

// ================================================================ host
extern "C" void kernel_launch(void* const* d_in, const int* in_sizes, int n_in,
                              void* d_out, int out_size, void* d_ws, size_t ws_size,
                              hipStream_t stream) {
    const float* style     = (const float*)d_in[0];
    const float* content   = (const float*)d_in[2];
    const float* pos_emb_s = (const float*)d_in[4];
    const float* newps_w   = (const float*)d_in[5];
    const float* newps_b   = (const float*)d_in[6];
    const float* vln_w     = (const float*)d_in[7];
    const float* vln_b     = (const float*)d_in[8];
    const float* vin_w     = (const float*)d_in[9];
    const float* vin_b     = (const float*)d_in[10];
    const float* vconv_w   = (const float*)d_in[11];
    const float* vconv_b   = (const float*)d_in[12];
    const float* vxproj_w  = (const float*)d_in[13];
    const float* vdt_bias  = (const float*)d_in[14];
    const float* vA_log    = (const float*)d_in[15];
    const float* vD        = (const float*)d_in[16];
    const float* von_w     = (const float*)d_in[17];
    const float* von_b     = (const float*)d_in[18];
    const float* vout_w    = (const float*)d_in[19];
    const float* vout_b    = (const float*)d_in[20];
    const float* lin1_w    = (const float*)d_in[21];
    const float* lin1_b    = (const float*)d_in[22];
    const float* lin2_w    = (const float*)d_in[23];
    const float* lin2_b    = (const float*)d_in[24];
    const float* dn1_w     = (const float*)d_in[25];
    const float* dn1_b     = (const float*)d_in[26];
    const float* dn2_w     = (const float*)d_in[27];
    const float* dn2_b     = (const float*)d_in[28];
    const float* dn3_w     = (const float*)d_in[29];
    const float* dn3_b     = (const float*)d_in[30];
    const float* decn_w    = (const float*)d_in[31];
    const float* decn_b    = (const float*)d_in[32];

    size_t off = 0;
    auto alloc = [&](size_t bytes) {
        void* p = (char*)d_ws + off;
        off += (bytes + 255) & ~(size_t)255;
        return p;
    };
    _Float16* zpad    = (_Float16*)alloc(256);
    _Float16* h_vin   = (_Float16*)alloc((size_t)4 * 512 * 512 * 2);
    _Float16* h_xproj = (_Float16*)alloc((size_t)4 * 544 * 512 * 2);
    _Float16* h_vout  = (_Float16*)alloc((size_t)4 * 512 * 512 * 2);
    _Float16* h_lin1  = (_Float16*)alloc((size_t)2048 * 512 * 2);
    _Float16* h_lin2  = (_Float16*)alloc((size_t)512 * 2048 * 2);
    _Float16* h_newps = (_Float16*)alloc((size_t)512 * 512 * 2);
    _Float16* pooled_h= (_Float16*)alloc((size_t)1296 * 512 * 2);
    float* posg   = (float*)alloc((size_t)1296 * 512 * 4);
    float* pc_seq = (float*)alloc((size_t)1024 * 512 * 4);
    float* ps_seq = (float*)alloc((size_t)1024 * 512 * 4);
    float* s_seq  = (float*)alloc((size_t)1024 * 512 * 4);
    float* c_seq  = (float*)alloc((size_t)1024 * 512 * 4);
    float* q_seq  = (float*)alloc((size_t)1024 * 512 * 4);
    float* k_seq  = (float*)alloc((size_t)1024 * 512 * 4);
    float* tgt    = (float*)alloc((size_t)1024 * 512 * 4);
    float* tgt2   = (float*)alloc((size_t)1024 * 512 * 4);
    float* ubuf   = (float*)alloc((size_t)1024 * 512 * 4);
    float* vbuf   = (float*)alloc((size_t)1024 * 512 * 4);
    float* ycomb  = (float*)alloc((size_t)1024 * 512 * 4);
    float* xdbl01 = (float*)alloc((size_t)1024 * 272 * 4);
    float* xdbl23 = (float*)alloc((size_t)1024 * 272 * 4);
    float* ysbuf  = (float*)alloc((size_t)4 * 1024 * 512 * 4);
    _Float16* xn_h   = (_Float16*)alloc((size_t)1024 * 512 * 2);
    _Float16* v_h    = (_Float16*)alloc((size_t)1024 * 512 * 2);
    _Float16* vflip_h= (_Float16*)alloc((size_t)1024 * 512 * 2);
    _Float16* tgt_h  = (_Float16*)alloc((size_t)1024 * 512 * 2);
    _Float16* hid_h  = (_Float16*)alloc((size_t)1024 * 2048 * 2);
    (void)ws_size; (void)in_sizes; (void)n_in; (void)out_size;

    zero_kernel<<<1, 256, 0, stream>>>(zpad, 128);

    auto cvt = [&](const float* s, _Float16* d, int n) {
        cvt_f16_kernel<<<(n + 255) / 256, 256, 0, stream>>>(s, d, n);
    };
    cvt(vin_w,    h_vin,   4 * 512 * 512);
    cvt(vxproj_w, h_xproj, 4 * 544 * 512);
    cvt(vout_w,   h_vout,  4 * 512 * 512);
    cvt(lin1_w,   h_lin1,  2048 * 512);
    cvt(lin2_w,   h_lin2,  512 * 2048);
    cvt(newps_w,  h_newps, 512 * 512);

    auto gemm = [&](const _Float16* A, const _Float16* W, const float* bias,
                    const float* res, float* of, _Float16* oh,
                    int M, int N, int K, int act) {
        dim3 g((M + 63) / 64, (N + 63) / 64);
        gemm_kernel<<<g, 256, 0, stream>>>(A, W, bias, res, of, oh, zpad, M, N, K, act);
    };

    // positional branch: pool -> newps GEMM -> bilinear resize -> pc_seq
    pool_kernel<<<1296, 256, 0, stream>>>(content, pooled_h);
    gemm(pooled_h, h_newps, newps_b, nullptr, posg, nullptr, 1296, 512, 512, 0);
    resize_kernel<<<1024, 256, 0, stream>>>(posg, pc_seq);

    toseq_kernel<<<2048, 256, 0, stream>>>(style, s_seq);
    toseq_kernel<<<2048, 256, 0, stream>>>(content, c_seq);
    toseq_kernel<<<2048, 256, 0, stream>>>(pos_emb_s, ps_seq);

    auto run_vss = [&](float* xbuf, int i) {
        ln_kernel<<<1024, 256, 0, stream>>>(xbuf, nullptr, vln_w + i * 512, vln_b + i * 512,
                                            nullptr, xn_h, 0);
        gemm(xn_h, h_vin + (size_t)i * 512 * 512, vin_b + i * 512, nullptr, ubuf, nullptr,
             1024, 512, 512, 0);
        conv_gelu_kernel<<<256, 256, 0, stream>>>(ubuf, vconv_w + i * 512 * 9,
                                                  vconv_b + i * 512, vbuf, v_h, vflip_h);
        gemm(v_h,     h_xproj + (size_t)i * 544 * 512,             nullptr, nullptr,
             xdbl01, nullptr, 1024, 272, 512, 0);
        gemm(vflip_h, h_xproj + (size_t)i * 544 * 512 + 272 * 512, nullptr, nullptr,
             xdbl23, nullptr, 1024, 272, 512, 0);
        scan_kernel<<<1024, 256, 0, stream>>>(xdbl01, xdbl23, vbuf, vdt_bias + i * 32,
                                              vA_log + i * 32, vD + i * 32, ysbuf);
        combine_kernel<<<2048, 256, 0, stream>>>(ysbuf, ycomb);
        ln_kernel<<<1024, 256, 0, stream>>>(ycomb, nullptr, von_w + i * 512, von_b + i * 512,
                                            nullptr, xn_h, 0);
        gemm(xn_h, h_vout + (size_t)i * 512 * 512, vout_b + i * 512, xbuf, xbuf, nullptr,
             1024, 512, 512, 0);
    };

    run_vss(s_seq, 0);                                   // s_enc
    run_vss(c_seq, 1);                                   // c_enc
    add_kernel<<<2048, 256, 0, stream>>>(c_seq, pc_seq, q_seq, 1024 * 512);
    add_kernel<<<2048, 256, 0, stream>>>(s_seq, ps_seq, k_seq, 1024 * 512);
    run_vss(q_seq, 2);                                   // vss(q)
    run_vss(k_seq, 3);                                   // vss(k)

    // tgt = LN(c_enc + vss(q)); tgt += LN(s_enc + vss(k)) (also emit f16 for FFN)
    ln_kernel<<<1024, 256, 0, stream>>>(q_seq, c_seq, dn1_w, dn1_b, tgt, nullptr, 0);
    ln_kernel<<<1024, 256, 0, stream>>>(k_seq, s_seq, dn2_w, dn2_b, tgt, tgt_h, 1);

    // FFN
    gemm(tgt_h, h_lin1, lin1_b, nullptr, nullptr, hid_h, 1024, 2048, 512, 1);
    gemm(hid_h, h_lin2, lin2_b, tgt, tgt2, nullptr, 1024, 512, 2048, 0);

    // double layernorm + output transpose
    ln_kernel<<<1024, 256, 0, stream>>>(tgt2, nullptr, dn3_w, dn3_b, tgt, nullptr, 0);
    ln_kernel<<<1024, 256, 0, stream>>>(tgt, nullptr, decn_w, decn_b, tgt2, nullptr, 0);
    outtr_kernel<<<2048, 256, 0, stream>>>(tgt2, (float*)d_out);
}
